// LocalWindowAttention_79147657330912
// MI455X (gfx1250) — matrix operand order repaired and checked by reading, not compiled, not run on hardware
//
#include <hip/hip_runtime.h>

typedef __attribute__((ext_vector_type(16))) _Float16 v16h;
typedef __attribute__((ext_vector_type(8)))  float    v8f;
typedef int v4i __attribute__((__vector_size__(4 * sizeof(int))));

// ---------------- problem constants ----------------
#define NHEAD 8
#define HDIM  32
#define TOK   64     // 8x8 window tokens
#define CDIM  256
#define IMG   256    // H == W == 256
#define NWIN  4096   // 4 * 32 * 32

// ---------------- async-to-LDS availability (compile-probe guarded) ----------
#if defined(__has_builtin)
#if __has_builtin(__builtin_amdgcn_global_load_async_to_lds_b128)
#define HAVE_ASYNC_LDS 1
#endif
#endif

#define ASG __attribute__((address_space(1)))
#define ASL __attribute__((address_space(3)))

// ---------------- LDS layout (bytes) ----------------
constexpr int XSTR   = 260;  // xs [t=64][c=256] f16 : B^T of QKV proj ([N][K])
constexpr int QKSTR  = 34;   // q/k [head][t=64][c=32] f16
constexpr int VSTR   = 66;   // v  [head][c=32][t=64] f16 : B^T of P*V^T
constexpr int PSTR   = 66;   // ps [head][i=64][j=64] f16 : A of P*V^T
constexpr int OSTR   = 260;  // ot [t=64][oc=256] f16 : B^T of Wo proj
constexpr int WF32STR = 36;  // weight chunk [o=256][k=32] f32 (pad 4: conflict-free, 16B-aligned)
constexpr int CHUNK_F32 = CDIM * WF32STR;        // one chunk, floats

constexpr int OFF_XS = 0;
constexpr int OFF_Q  = OFF_XS + TOK * XSTR * 2;
constexpr int OFF_K  = OFF_Q  + NHEAD * TOK * QKSTR * 2;
constexpr int OFF_V  = OFF_K  + NHEAD * TOK * QKSTR * 2;
constexpr int OFF_P  = OFF_V  + NHEAD * HDIM * VSTR * 2;
constexpr int OFF_O  = OFF_P  + NHEAD * TOK * PSTR * 2;
constexpr int OFF_WB = OFF_O  + TOK * OSTR * 2;                  // 16B aligned
constexpr int LDS_BYTES = OFF_WB + 2 * CHUNK_F32 * 4;            // ~304 KB <= 320 KB

// ---------------- WMMA helpers ----------------
__device__ __forceinline__ v8f zero8() {
  v8f z = {0.f, 0.f, 0.f, 0.f, 0.f, 0.f, 0.f, 0.f};
  return z;
}

__device__ __forceinline__ v8f wmma_f16(v16h a, v16h b, v8f c) {
  return __builtin_amdgcn_wmma_f32_16x16x32_f16(false, a, false, b, (short)0, c,
                                                false, false);
}

// A fragment from f32 LDS source, row-major M x K, ld in floats; converts to f16.
// ISA 7.12.2 16-bit A 16x32: lane holds row m=l&15; g=l>>4;
// vgpr r, slot s -> k = 8g + 16*(r>>2) + 2*(r&3) + s
__device__ __forceinline__ v16h frag_a_f32(const float* __restrict__ s, int ld) {
  const int lane = threadIdx.x & 31;
  const int g = lane >> 4, m = lane & 15;
  v16h f;
#pragma unroll
  for (int r = 0; r < 8; ++r) {
    const int k = 8 * g + 16 * (r >> 2) + 2 * (r & 3);
    f[2 * r]     = (_Float16)s[m * ld + k];
    f[2 * r + 1] = (_Float16)s[m * ld + k + 1];
  }
  return f;
}

// A fragment from f16 row-major M x K source, ld in halfs.
__device__ __forceinline__ v16h frag_a(const _Float16* __restrict__ s, int ld) {
  const int lane = threadIdx.x & 31;
  const int g = lane >> 4, m = lane & 15;
  v16h f;
#pragma unroll
  for (int r = 0; r < 8; ++r) {
    const int k = 8 * g + 16 * (r >> 2) + 2 * (r & 3);
    f[2 * r]     = s[m * ld + k];
    f[2 * r + 1] = s[m * ld + k + 1];
  }
  return f;
}

// B fragment from transposed (N x K) f16 source, ld in halfs.
// lane holds column n=l&15; element h -> K = 8g + (h&7) + 16*(h>>3) (matches A).
__device__ __forceinline__ v16h frag_bt(const _Float16* __restrict__ s, int ld) {
  const int lane = threadIdx.x & 31;
  const int g = lane >> 4, n = lane & 15;
  v16h f;
#pragma unroll
  for (int h = 0; h < 16; ++h) f[h] = s[n * ld + 8 * g + (h & 7) + 16 * (h >> 3)];
  return f;
}

// ---------------- weight-chunk streaming (double buffered) ----------------
#if defined(HAVE_ASYNC_LDS)
__device__ __forceinline__ void chunk_start(const float* __restrict__ W, int k0,
                                            float* dst, int wrow, int wcol) {
  // per thread: 8 x 16B async copies, fully coalesced (8 lanes cover one 128B row seg)
#pragma unroll
  for (int p = 0; p < 8; ++p) {
    const float* g = &W[(size_t)(p * 32 + wrow) * CDIM + (size_t)k0 * 32 + wcol];
    float* l = &dst[(p * 32 + wrow) * WF32STR + wcol];
    __builtin_amdgcn_global_load_async_to_lds_b128(
        (ASG v4i*)(__UINTPTR_TYPE__)g,
        (ASL v4i*)(unsigned)(__UINTPTR_TYPE__)l, 0, 0);
  }
}
__device__ __forceinline__ void chunk_finish(float4* /*regs*/, float* /*cur*/,
                                             int /*wrow*/, int /*wcol*/) {
#if __has_builtin(__builtin_amdgcn_s_wait_asynccnt)
  __builtin_amdgcn_s_wait_asynccnt(0);
#else
  asm volatile("s_wait_asynccnt 0" ::: "memory");
#endif
}
#else
__device__ __forceinline__ void chunk_start(const float* __restrict__ W, int k0,
                                            float4* regs, int wrow, int wcol) {
#pragma unroll
  for (int p = 0; p < 8; ++p)
    regs[p] = *(const float4*)&W[(size_t)(p * 32 + wrow) * CDIM +
                                 (size_t)k0 * 32 + wcol];
}
__device__ __forceinline__ void chunk_finish(float4* regs, float* cur, int wrow,
                                             int wcol) {
#pragma unroll
  for (int p = 0; p < 8; ++p)
    *(float4*)&cur[(p * 32 + wrow) * WF32STR + wcol] = regs[p];
}
#endif

// ---------------- one 256x64x256 GEMM pass (acc += W * B) ----------------
__device__ __forceinline__ void gemm256(const float* __restrict__ W,
                                        const _Float16* __restrict__ Bsrc, int bld,
                                        float* __restrict__ wbuf, int tid, int wave,
                                        v8f (&acc)[2][4]) {
  const int wrow = tid >> 3;         // 0..31: row within a 32-row group
  const int wcol = (tid & 7) << 2;   // float4 column
  float4 regs[8];
#if defined(HAVE_ASYNC_LDS)
  chunk_start(W, 0, wbuf, wrow, wcol);
#else
  chunk_start(W, 0, regs, wrow, wcol);
#endif
  for (int k0 = 0; k0 < 8; ++k0) {
    float* cur = wbuf + (k0 & 1) * CHUNK_F32;
    chunk_finish(regs, cur, wrow, wcol);   // async: wait own copies; sync: store regs
    __syncthreads();                       // publish chunk k0 to all waves
    if (k0 < 7) {
      float* nxt = wbuf + ((k0 + 1) & 1) * CHUNK_F32;
#if defined(HAVE_ASYNC_LDS)
      chunk_start(W, k0 + 1, nxt, wrow, wcol);
#else
      chunk_start(W, k0 + 1, regs, wrow, wcol);
      (void)nxt;
#endif
    }
    v16h bfr[4];
#pragma unroll
    for (int nt = 0; nt < 4; ++nt)
      bfr[nt] = frag_bt(Bsrc + nt * 16 * bld + k0 * 32, bld);
#pragma unroll
    for (int mt = 0; mt < 2; ++mt) {
      const v16h afr = frag_a_f32(cur + (wave * 32 + mt * 16) * WF32STR, WF32STR);
#pragma unroll
      for (int nt = 0; nt < 4; ++nt)
        acc[mt][nt] = wmma_f16(afr, bfr[nt], acc[mt][nt]);
    }
  }
}

// ---------------- fused local-window attention ----------------
__global__ void __launch_bounds__(256)
win_attn_fused(const float* __restrict__ x,
               const float* __restrict__ Wq,
               const float* __restrict__ Wk,
               const float* __restrict__ Wv,
               const float* __restrict__ Wo,
               float* __restrict__ out) {
  extern __shared__ char smraw[];
  _Float16* xs  = (_Float16*)(smraw + OFF_XS);
  _Float16* qls = (_Float16*)(smraw + OFF_Q);
  _Float16* kls = (_Float16*)(smraw + OFF_K);
  _Float16* vls = (_Float16*)(smraw + OFF_V);
  _Float16* ps  = (_Float16*)(smraw + OFF_P);
  _Float16* ot  = (_Float16*)(smraw + OFF_O);
  float*    wbuf = (float*)(smraw + OFF_WB);

  const int tid  = threadIdx.x;
  const int lane = tid & 31;
  const int wave = tid >> 5;  // wave == head == M-slice [32w, 32w+32)
  const int g    = lane >> 4;
  const int n    = lane & 15;

  const int win = blockIdx.x;
  const int b   = win >> 10;
  const int h0  = (win >> 5) & 31;
  const int w0  = win & 31;

  const size_t pixBase =
      (size_t)b * CDIM * IMG * IMG + (size_t)(h0 * 8) * IMG + (size_t)(w0 * 8);

  // ---- stage x window into LDS as f16, [t][c] (float4 loads, 2 per 8-px row) ----
#pragma unroll
  for (int p = 0; p < 8; ++p) {
    const int rowIdx = p * 256 + tid;  // 2048 (c,i) rows
    const int c = rowIdx >> 3, i = rowIdx & 7;
    const float* rp = &x[pixBase + (size_t)c * (IMG * IMG) + i * IMG];
    const float4 u0 = *(const float4*)rp;
    const float4 u1 = *(const float4*)(rp + 4);
    const int t0 = i * 8;
    xs[(t0 + 0) * XSTR + c] = (_Float16)u0.x;
    xs[(t0 + 1) * XSTR + c] = (_Float16)u0.y;
    xs[(t0 + 2) * XSTR + c] = (_Float16)u0.z;
    xs[(t0 + 3) * XSTR + c] = (_Float16)u0.w;
    xs[(t0 + 4) * XSTR + c] = (_Float16)u1.x;
    xs[(t0 + 5) * XSTR + c] = (_Float16)u1.y;
    xs[(t0 + 6) * XSTR + c] = (_Float16)u1.z;
    xs[(t0 + 7) * XSTR + c] = (_Float16)u1.w;
  }
  __syncthreads();

  const float scale = 0.17677669529663687f;  // 32^-0.5, folded into Q

  // ---- Q projection ----
  {
    v8f acc[2][4];
#pragma unroll
    for (int mt = 0; mt < 2; ++mt)
#pragma unroll
      for (int nt = 0; nt < 4; ++nt) acc[mt][nt] = zero8();
    gemm256(Wq, xs, XSTR, wbuf, tid, wave, acc);
#pragma unroll
    for (int mt = 0; mt < 2; ++mt)
#pragma unroll
      for (int nt = 0; nt < 4; ++nt)
#pragma unroll
        for (int r = 0; r < 8; ++r) {
          const int cl = mt * 16 + r + 8 * g;
          const int t  = nt * 16 + n;
          qls[(wave * TOK + t) * QKSTR + cl] = (_Float16)(acc[mt][nt][r] * scale);
        }
  }
  // ---- K projection ----
  {
    v8f acc[2][4];
#pragma unroll
    for (int mt = 0; mt < 2; ++mt)
#pragma unroll
      for (int nt = 0; nt < 4; ++nt) acc[mt][nt] = zero8();
    gemm256(Wk, xs, XSTR, wbuf, tid, wave, acc);
#pragma unroll
    for (int mt = 0; mt < 2; ++mt)
#pragma unroll
      for (int nt = 0; nt < 4; ++nt)
#pragma unroll
        for (int r = 0; r < 8; ++r) {
          const int cl = mt * 16 + r + 8 * g;
          const int t  = nt * 16 + n;
          kls[(wave * TOK + t) * QKSTR + cl] = (_Float16)acc[mt][nt][r];
        }
  }
  // ---- V projection ----
  {
    v8f acc[2][4];
#pragma unroll
    for (int mt = 0; mt < 2; ++mt)
#pragma unroll
      for (int nt = 0; nt < 4; ++nt) acc[mt][nt] = zero8();
    gemm256(Wv, xs, XSTR, wbuf, tid, wave, acc);
#pragma unroll
    for (int mt = 0; mt < 2; ++mt)
#pragma unroll
      for (int nt = 0; nt < 4; ++nt)
#pragma unroll
        for (int r = 0; r < 8; ++r) {
          const int cl = mt * 16 + r + 8 * g;
          const int t  = nt * 16 + n;
          vls[(wave * HDIM + cl) * VSTR + t] = (_Float16)acc[mt][nt][r];
        }
  }
  // No barrier needed: wave w produced exactly head w's Q/K/V.

  // ---- attention, one head per wave ----
  const _Float16* qh = qls + wave * TOK * QKSTR;
  const _Float16* kh = kls + wave * TOK * QKSTR;
  const _Float16* vh = vls + wave * HDIM * VSTR;
  _Float16* ph = ps + wave * TOK * PSTR;

  v8f sacc[4][4];
#pragma unroll
  for (int mi = 0; mi < 4; ++mi)
#pragma unroll
    for (int nj = 0; nj < 4; ++nj) sacc[mi][nj] = zero8();

  v16h aq[4];
#pragma unroll
  for (int mi = 0; mi < 4; ++mi) aq[mi] = frag_a(qh + mi * 16 * QKSTR, QKSTR);
#pragma unroll
  for (int nj = 0; nj < 4; ++nj) {
    const v16h bk = frag_bt(kh + nj * 16 * QKSTR, QKSTR);
#pragma unroll
    for (int mi = 0; mi < 4; ++mi)
      sacc[mi][nj] = wmma_f16(aq[mi], bk, sacc[mi][nj]);
  }

  // softmax over rows i = mi*16 + r + 8g (row spans 16 lanes of one half-wave)
#pragma unroll
  for (int mi = 0; mi < 4; ++mi)
#pragma unroll
    for (int r = 0; r < 8; ++r) {
      float mx = -1e30f;
#pragma unroll
      for (int nj = 0; nj < 4; ++nj) mx = fmaxf(mx, sacc[mi][nj][r]);
#pragma unroll
      for (int off = 8; off >= 1; off >>= 1)
        mx = fmaxf(mx, __shfl_xor(mx, off, 32));
      float sum = 0.f;
#pragma unroll
      for (int nj = 0; nj < 4; ++nj) {
        const float e = __expf(sacc[mi][nj][r] - mx);
        sacc[mi][nj][r] = e;
        sum += e;
      }
#pragma unroll
      for (int off = 8; off >= 1; off >>= 1) sum += __shfl_xor(sum, off, 32);
      const float inv = 1.f / sum;
      const int i = mi * 16 + r + 8 * g;
#pragma unroll
      for (int nj = 0; nj < 4; ++nj)
        ph[i * PSTR + nj * 16 + n] = (_Float16)(sacc[mi][nj][r] * inv);
    }

  // O^T = P * V^T : M=i(4 tiles), N=c(2 tiles), K=j(2 steps of 32)
  v8f oacc[4][2];
#pragma unroll
  for (int mi = 0; mi < 4; ++mi)
#pragma unroll
    for (int nc = 0; nc < 2; ++nc) oacc[mi][nc] = zero8();

#pragma unroll
  for (int ks = 0; ks < 2; ++ks) {
    v16h ap[4];
#pragma unroll
    for (int mi = 0; mi < 4; ++mi)
      ap[mi] = frag_a(ph + mi * 16 * PSTR + ks * 32, PSTR);
#pragma unroll
    for (int nc = 0; nc < 2; ++nc) {
      const v16h bv = frag_bt(vh + nc * 16 * VSTR + ks * 32, VSTR);
#pragma unroll
      for (int mi = 0; mi < 4; ++mi)
        oacc[mi][nc] = wmma_f16(ap[mi], bv, oacc[mi][nc]);
    }
  }

  // store window output ot[t][head*32 + c]  (B^T layout for Wo GEMM)
#pragma unroll
  for (int mi = 0; mi < 4; ++mi)
#pragma unroll
    for (int nc = 0; nc < 2; ++nc)
#pragma unroll
      for (int r = 0; r < 8; ++r) {
        const int t = mi * 16 + r + 8 * g;
        const int c = nc * 16 + n;
        ot[t * OSTR + wave * 32 + c] = (_Float16)oacc[mi][nc][r];
      }

  // ---- output projection: out = Wo * O (gemm's first barrier publishes ot) ----
  {
    v8f facc[2][4];
#pragma unroll
    for (int mt = 0; mt < 2; ++mt)
#pragma unroll
      for (int nt = 0; nt < 4; ++nt) facc[mt][nt] = zero8();
    gemm256(Wo, ot, OSTR, wbuf, tid, wave, facc);

#pragma unroll
    for (int mt = 0; mt < 2; ++mt)
#pragma unroll
      for (int nt = 0; nt < 4; ++nt)
#pragma unroll
        for (int r = 0; r < 8; ++r) {
          const int o = wave * 32 + mt * 16 + r + 8 * g;
          const int t = nt * 16 + n;
          const int i = t >> 3, j = t & 7;
          out[pixBase + (size_t)o * (IMG * IMG) + i * IMG + j] = facc[mt][nt][r];
        }
  }
}

// ---------------- host launch ----------------
extern "C" void kernel_launch(void* const* d_in, const int* in_sizes, int n_in,
                              void* d_out, int out_size, void* d_ws, size_t ws_size,
                              hipStream_t stream) {
  (void)in_sizes; (void)n_in; (void)out_size; (void)d_ws; (void)ws_size;
  const float* x  = (const float*)d_in[0];
  const float* Wq = (const float*)d_in[1];
  const float* Wk = (const float*)d_in[2];
  const float* Wv = (const float*)d_in[3];
  const float* Wo = (const float*)d_in[4];
  float* out = (float*)d_out;

  (void)hipFuncSetAttribute((const void*)win_attn_fused,
                            hipFuncAttributeMaxDynamicSharedMemorySize, LDS_BYTES);
  win_attn_fused<<<dim3(NWIN), dim3(256), LDS_BYTES, stream>>>(x, Wq, Wk, Wv, Wo, out);
}
